// EfficientAttention_7859790152139
// MI455X (gfx1250) — matrix operand order repaired and checked by reading, not compiled, not run on hardware
//
#include <hip/hip_runtime.h>

// Problem dims
#define BATCH 16
#define NSEQ  8192
#define DFEAT 256
#define TM    32      // rows per tile
#define SPLIT 16      // workgroups per batch (persistent, register ctx accumulation)
#define XS_STR 264    // bf16 elems per Xs row (256 + 8 pad -> conflict-free frag loads)
#define T_STR  40     // bf16 elems per Et/Vt row (32 + 8 pad)
#define LOG2E 1.4426950408889634f

typedef __attribute__((ext_vector_type(16))) __bf16 v16bf;
typedef __attribute__((ext_vector_type(2)))  __bf16 v2bf;
typedef __attribute__((ext_vector_type(2)))  float  v2f;
typedef __attribute__((ext_vector_type(8)))  float  v8f;

union Frag {
  v16bf v;
  uint4 q[2];
};

// Pack two f32 -> packed bf16 pair; paired conversion selects a single
// v_cvt_pk_bf16_f32 with both operands (no shift/or re-packing).
__device__ __forceinline__ unsigned int pack2(float lo, float hi) {
  v2f t = {lo, hi};
  v2bf r = __builtin_convertvector(t, v2bf);
  return __builtin_bit_cast(unsigned int, r);
}

__device__ __forceinline__ unsigned short to_bf16_bits(float f) {
  union { __bf16 h; unsigned short u; } cv;
  cv.h = (__bf16)f;
  return cv.u;
}

// exp2 -> v_exp_f32 directly
__device__ __forceinline__ float fast_exp2(float x) {
#if __has_builtin(__builtin_amdgcn_exp2f)
  return __builtin_amdgcn_exp2f(x);
#else
  return exp2f(x);
#endif
}

__device__ __forceinline__ void gadd(float* p, float v) {
  __hip_atomic_fetch_add(p, v, __ATOMIC_RELAXED, __HIP_MEMORY_SCOPE_AGENT);
}

__device__ __forceinline__ v8f wmma_bf16(v16bf a, v16bf b, v8f c) {
  // D = A(16x32 bf16) * B(32x16 bf16) + C(16x16 f32)
  return __builtin_amdgcn_wmma_f32_16x16x32_bf16(false, a, false, b, (short)0, c,
                                                 false, false);
}

// One 16x16 projection tile. A fragments re-read from LDS per WMMA.
// kf loop unrolled only 2x: the loop back-edge fences the scheduler so it
// cannot hoist all 8 global B-fragment loads (which previously spilled).
__device__ __forceinline__ v8f proj_tile(const unsigned short* xrow,  // &Xs[m*XS_STR]
                                         const unsigned short* __restrict__ wT,
                                         int n, int hh, float bias) {
  v8f acc = {bias, bias, bias, bias, bias, bias, bias, bias};
  const unsigned short* wbase = wT + n * 256 + 8 * hh;
#pragma unroll 2
  for (int kf = 0; kf < 8; ++kf) {
    Frag a_, b_;
    const unsigned short* xp = xrow + kf * 32 + 8 * hh;
    a_.q[0] = *(const uint4*)xp;          // ds_load_b128
    a_.q[1] = *(const uint4*)(xp + 16);   // ds_load_b128
    const unsigned short* wp = wbase + kf * 32;
    b_.q[0] = *(const uint4*)wp;          // global_load_b128 (L2-resident weights)
    b_.q[1] = *(const uint4*)(wp + 16);
    acc = wmma_bf16(a_.v, b_.v, acc);
  }
  return acc;
}

// ---------------- prelude: W[k][n] f32 -> WT[n][k] bf16 (optionally * log2e) ----
__global__ __launch_bounds__(256) void ea_wt(const float* __restrict__ w,
                                             unsigned short* __restrict__ wt,
                                             float scale) {
  int n = blockIdx.x, k = threadIdx.x;
  wt[n * 256 + k] = to_bf16_bits(w[k * 256 + n] * scale);
}

__global__ __launch_bounds__(256) void ea_zero(float* __restrict__ p, int n) {
  int i = blockIdx.x * 256 + threadIdx.x;
  if (i < n) p[i] = 0.0f;
}

// ---------------- main fused kernel ----------------
__global__ __launch_bounds__(256)
__attribute__((amdgpu_waves_per_eu(2))) void ea_main(
    const float* __restrict__ x,
    const unsigned short* __restrict__ wqT, const unsigned short* __restrict__ wkT,
    const unsigned short* __restrict__ wvT,
    const float* __restrict__ bq, const float* __restrict__ bk,
    const float* __restrict__ bv,
    float* __restrict__ qsum, float* __restrict__ colsum,
    float* __restrict__ ctxPart) {
  __shared__ __align__(16) unsigned short Xs[TM * XS_STR];    // 16896 B
  __shared__ __align__(16) unsigned short Et[256 * T_STR];    // 20480 B (exp(Q), then E^T)
  __shared__ __align__(16) unsigned short Vt[256 * T_STR];    // 20480 B (V^T bf16)
  __shared__ float rowsumL[TM];

  const int b    = blockIdx.x / SPLIT;
  const int s    = blockIdx.x % SPLIT;
  const int tid  = threadIdx.x;
  const int w    = tid >> 5;
  const int lane = tid & 31;
  const int hh   = lane >> 4;    // wave half
  const int n16  = lane & 15;
  const int band = w & 1;        // 16-row band within 32-row tile
  const int ctg  = w >> 1;       // owns col-tiles ctg*4 .. ctg*4+3 for projections
  const int vt0  = w * 2;        // owns v-tiles vt0, vt0+1 for ctx GEMM

  const unsigned short* xrow = &Xs[(band * 16 + n16) * XS_STR];
  const int packIdx = band * 16 + hh * 8;  // column offset for packed C-frag stores

  // per-batch ctx accumulator slice lives in registers: 2 vtiles x 16 ctiles
  v8f cacc[2][16];
  const v8f vzero = {0, 0, 0, 0, 0, 0, 0, 0};
#pragma unroll
  for (int i = 0; i < 2; ++i)
#pragma unroll
    for (int j = 0; j < 16; ++j) cacc[i][j] = vzero;

  const int tiles = NSEQ / (TM * SPLIT);  // 16
  for (int it = 0; it < tiles; ++it) {
    const int n0 = (s * tiles + it) * TM;
    __syncthreads();  // (a) protect LDS from previous iteration's readers

    // ---- stage X tile (f32 global -> bf16 LDS), coalesced ----
    {
      const int r  = tid >> 3;
      const int c0 = (tid & 7) * 32;
      const float* src = x + ((size_t)b * NSEQ + n0 + r) * DFEAT + c0;
      uint4* dst = (uint4*)&Xs[r * XS_STR + c0];
#pragma unroll
      for (int j = 0; j < 4; ++j) {
        float4 f0 = ((const float4*)src)[2 * j];
        float4 f1 = ((const float4*)src)[2 * j + 1];
        uint4 o;
        o.x = pack2(f0.x, f0.y);
        o.y = pack2(f0.z, f0.w);
        o.z = pack2(f1.x, f1.y);
        o.w = pack2(f1.z, f1.w);
        dst[j] = o;
      }
      if (it + 1 < tiles) __builtin_prefetch(src + TM * DFEAT, 0, 1);
      if (tid < TM) rowsumL[tid] = 0.0f;
    }
    __syncthreads();  // (b)

    // ---- Q projection (logits pre-scaled by log2e): exp2 -> Et, rowsums ----
    {
      float rsum[8] = {0, 0, 0, 0, 0, 0, 0, 0};
#pragma unroll
      for (int ctl = 0; ctl < 4; ++ctl) {
        const int n = (ctg * 4 + ctl) * 16 + n16;
        v8f acc = proj_tile(xrow, wqT, n, hh, bq[n] * LOG2E);
        float e0[8];
#pragma unroll
        for (int r = 0; r < 8; ++r) {
          e0[r] = fast_exp2(acc[r]);
          rsum[r] += e0[r];
        }
        uint4 o;
        o.x = pack2(e0[0], e0[1]);
        o.y = pack2(e0[2], e0[3]);
        o.z = pack2(e0[4], e0[5]);
        o.w = pack2(e0[6], e0[7]);
        *(uint4*)&Et[n * T_STR + packIdx] = o;  // ds_store_b128
      }
#pragma unroll
      for (int r = 0; r < 8; ++r)
        atomicAdd(&rowsumL[packIdx + r], rsum[r]);  // ds_add_f32
    }
    __syncthreads();  // (c) rowsums + parked exp(Q) visible

    {  // normalize and accumulate per-column softmax sums
      float inv[8];
#pragma unroll
      for (int r = 0; r < 8; ++r) inv[r] = 1.0f / rowsumL[packIdx + r];
#pragma unroll
      for (int ctl = 0; ctl < 4; ++ctl) {
        const int n = (ctg * 4 + ctl) * 16 + n16;
        const uint4 o = *(const uint4*)&Et[n * T_STR + packIdx];
        float p = 0.0f;
        p += __uint_as_float(o.x << 16) * inv[0];
        p += __uint_as_float(o.x & 0xffff0000u) * inv[1];
        p += __uint_as_float(o.y << 16) * inv[2];
        p += __uint_as_float(o.y & 0xffff0000u) * inv[3];
        p += __uint_as_float(o.z << 16) * inv[4];
        p += __uint_as_float(o.z & 0xffff0000u) * inv[5];
        p += __uint_as_float(o.w << 16) * inv[6];
        p += __uint_as_float(o.w & 0xffff0000u) * inv[7];
        gadd(&qsum[b * 256 + n], p);
      }
    }
    __syncthreads();  // (c2) Q readers done before K overwrites Et

    // ---- K projection (pre-scaled) -> E = exp(K): colsum + E^T bf16 ----
#pragma unroll
    for (int ctl = 0; ctl < 4; ++ctl) {
      const int n = (ctg * 4 + ctl) * 16 + n16;
      v8f acc = proj_tile(xrow, wkT, n, hh, bk[n] * LOG2E);
      float e0[8], cs = 0.0f;
#pragma unroll
      for (int r = 0; r < 8; ++r) {
        e0[r] = fast_exp2(acc[r]);
        cs += e0[r];
      }
      gadd(&colsum[b * 256 + n], cs);
      uint4 o;
      o.x = pack2(e0[0], e0[1]);
      o.y = pack2(e0[2], e0[3]);
      o.z = pack2(e0[4], e0[5]);
      o.w = pack2(e0[6], e0[7]);
      *(uint4*)&Et[n * T_STR + packIdx] = o;
    }

    // ---- V projection -> V^T bf16 into LDS ----
#pragma unroll
    for (int ctl = 0; ctl < 4; ++ctl) {
      const int n = (ctg * 4 + ctl) * 16 + n16;
      v8f acc = proj_tile(xrow, wvT, n, hh, bv[n]);
      uint4 o;
      o.x = pack2(acc[0], acc[1]);
      o.y = pack2(acc[2], acc[3]);
      o.z = pack2(acc[4], acc[5]);
      o.w = pack2(acc[6], acc[7]);
      *(uint4*)&Vt[n * T_STR + packIdx] = o;
    }
    __syncthreads();  // (d) E^T / V^T complete

    // ---- ctx += E^T (256 x 32) @ V (32 x 256), K-dim = TM = 32 ----
    // Fully unrolled (cacc indices must stay constant); sched barriers stop
    // the scheduler from hoisting all 16 A-fragment loads over the WMMAs.
#pragma unroll
    for (int vtl = 0; vtl < 2; ++vtl) {
      const int vrow = (vt0 + vtl) * 16 + n16;  // row of V^T = v-feature
      Frag bv_;
      const unsigned short* vp = &Vt[vrow * T_STR + 16 * hh];
      bv_.q[0] = *(const uint4*)vp;
      bv_.q[1] = *(const uint4*)(vp + 8);
#pragma unroll
      for (int ct = 0; ct < 16; ++ct) {
        Frag ea_;
        const int c = ct * 16 + n16;  // row of E^T = k-feature
        const unsigned short* ep = &Et[c * T_STR + 16 * hh];
        ea_.q[0] = *(const uint4*)ep;
        ea_.q[1] = *(const uint4*)(ep + 8);
        cacc[vtl][ct] = wmma_bf16(ea_.v, bv_.v, cacc[vtl][ct]);
        if ((ct & 7) == 7) __builtin_amdgcn_sched_barrier(0);
      }
    }
  }

  // ---- flush register ctx accumulators to this split's partial buffer ----
  float* dst = ctxPart + (size_t)(b * SPLIT + s) * (256 * 256);
#pragma unroll
  for (int vtl = 0; vtl < 2; ++vtl)
#pragma unroll
    for (int ct = 0; ct < 16; ++ct)
#pragma unroll
      for (int r = 0; r < 8; ++r) {
        const int c = ct * 16 + hh * 8 + r;
        const int v = (vt0 + vtl) * 16 + n16;
        dst[c * 256 + v] = cacc[vtl][ct][r];
      }
}

// ---------------- finalize: out[b][v] += (1/N) * sum_c (qsum/colsum)[c] * ctxPart[s][c][v]
// Parallel over (b, s); out pre-zeroed; 16 atomic adds per element.
__global__ __launch_bounds__(256) void ea_fin(const float* __restrict__ qsum,
                                              const float* __restrict__ colsum,
                                              const float* __restrict__ ctxPart,
                                              float* __restrict__ out) {
  __shared__ float wv[256];
  const int b = blockIdx.x / SPLIT;
  const int s = blockIdx.x % SPLIT;
  const int t = threadIdx.x;
  wv[t] = qsum[b * 256 + t] / colsum[b * 256 + t];
  __syncthreads();
  const float* cp = ctxPart + (size_t)(b * SPLIT + s) * (256 * 256);
  float acc = 0.0f;
#pragma unroll 4
  for (int c = 0; c < 256; ++c) acc += wv[c] * cp[c * 256 + t];
  gadd(&out[b * 256 + t], acc * (1.0f / (float)NSEQ));
}

extern "C" void kernel_launch(void* const* d_in, const int* in_sizes, int n_in,
                              void* d_out, int out_size, void* d_ws, size_t ws_size,
                              hipStream_t stream) {
  const float* x  = (const float*)d_in[0];
  const float* Wq = (const float*)d_in[1];
  const float* bq = (const float*)d_in[2];
  const float* Wk = (const float*)d_in[3];
  const float* bk = (const float*)d_in[4];
  const float* Wv = (const float*)d_in[5];
  const float* bv = (const float*)d_in[6];
  float* out = (float*)d_out;

  // workspace layout
  unsigned short* wqT = (unsigned short*)d_ws;          // 128 KB
  unsigned short* wkT = wqT + 256 * 256;                // 128 KB
  unsigned short* wvT = wkT + 256 * 256;                // 128 KB
  float* qsum   = (float*)(wvT + 256 * 256);            // 16 KB
  float* colsum = qsum + BATCH * 256;                   // 16 KB
  float* ctxPart = colsum + BATCH * 256;                // 64 MB (B*SPLIT*256*256 f32)

  // Wq / Wk pre-scaled by log2e so exp() becomes bare v_exp_f32 (exp2).
  ea_wt<<<256, 256, 0, stream>>>(Wq, wqT, LOG2E);
  ea_wt<<<256, 256, 0, stream>>>(Wk, wkT, LOG2E);
  ea_wt<<<256, 256, 0, stream>>>(Wv, wvT, 1.0f);
  ea_zero<<<(BATCH * 512 + 255) / 256, 256, 0, stream>>>(qsum, BATCH * 512);
  ea_zero<<<(BATCH * 256 + 255) / 256, 256, 0, stream>>>(out, BATCH * 256);

  ea_main<<<BATCH * SPLIT, 256, 0, stream>>>(x, wqT, wkT, wvT, bq, bk, bv,
                                             qsum, colsum, ctxPart);
  ea_fin<<<BATCH * SPLIT, 256, 0, stream>>>(qsum, colsum, ctxPart, out);
}
